// DecoderRNNNetwork_38826504356355
// MI455X (gfx1250) — matrix operand order repaired and checked by reading, not compiled, not run on hardware
//
#include <hip/hip_runtime.h>
#include <hip/hip_bf16.h>

typedef _Float16 v16h __attribute__((ext_vector_type(16)));
typedef _Float16 h8   __attribute__((ext_vector_type(8)));
typedef float    v8f  __attribute__((ext_vector_type(8)));

#define T_DIM 4096
#define B_DIM 32
#define K_EP  16
#define LM    256
#define S_DIM 128
#define C_DIM 32
#define M_DIM 32
#define E_S   128
#define E_C   64
#define H_DIM 256
#define G_DIM 768
#define A_DIM 16
#define ROWS  (T_DIM * B_DIM)   // 131072

// ---------------- WMMA fragment helpers (CDNA5 wave32, 16x16x32 f16) ----------

__device__ __forceinline__ v8f splat8(float v) {
  v8f r;
#pragma unroll
  for (int i = 0; i < 8; i++) r[i] = v;
  return r;
}

__device__ __forceinline__ v8f wmma16(v16h a, v16h b, v8f c) {
  // 8 args: (neg_a, A, neg_b, B, c_mod, C, reuse_a, reuse_b)
  return __builtin_amdgcn_wmma_f32_16x16x32_f16(false, a, false, b, (short)0, c,
                                                false, false);
}

// A-fragment (16x32 f16, row-major source, row stride in elements)
__device__ __forceinline__ v16h load_a_f16(const _Float16* base, int stride,
                                           int kt, int lane) {
  int m = lane & 15, off = (lane >> 4) * 8;
  const _Float16* p = base + m * stride + kt * 32 + off;
  h8 lo = *(const h8*)p;
  h8 hi = *(const h8*)(p + 16);
  v16h a;
#pragma unroll
  for (int i = 0; i < 8; i++) { a[i] = lo[i]; a[i + 8] = hi[i]; }
  return a;
}

// A-fragment from f32 source with on-the-fly conversion
__device__ __forceinline__ v16h load_a_f32(const float* base, int stride,
                                           int kt, int lane) {
  int m = lane & 15, off = (lane >> 4) * 8;
  const float* p = base + m * stride + kt * 32 + off;
  v16h a;
#pragma unroll
  for (int i = 0; i < 8; i++) {
    a[i]     = (_Float16)p[i];
    a[i + 8] = (_Float16)p[16 + i];
  }
  return a;
}

// pre-swizzled B-fragment: frag (kt,nt) = 32 lanes x 16 f16, lane-contiguous
__device__ __forceinline__ v16h load_b_frag(const _Float16* fragbase, int NT,
                                            int kt, int nt, int lane) {
  return *(const v16h*)(fragbase + (((size_t)kt * NT + nt) * 32 + lane) * 16);
}

// ---------------- Phase 0: weight swizzle into WMMA-B fragment layout --------

__global__ void swizzle_b_kernel(const float* __restrict__ W, int K, int N,
                                 _Float16* __restrict__ out, int KT, int NT) {
  int tid = blockIdx.x * blockDim.x + threadIdx.x;
  int total = KT * NT * 32;
  if (tid >= total) return;
  int lane = tid & 31;
  int fr = tid >> 5;
  int nt = fr % NT, kt = fr / NT;
  int n = nt * 16 + (lane & 15);
  int kbase = kt * 32 + (lane >> 4) * 8;
  _Float16* dst = out + ((size_t)fr * 32 + lane) * 16;
#pragma unroll
  for (int i = 0; i < 16; i++) {
    int k = kbase + (i < 8 ? i : i + 8);
    float v = (k < K && n < N) ? W[(size_t)k * N + n] : 0.f;
    dst[i] = (_Float16)v;
  }
}

__global__ void bias16_kernel(const float* __restrict__ b, int n,
                              _Float16* __restrict__ out) {
  int i = blockIdx.x * blockDim.x + threadIdx.x;
  if (i < n) out[i] = (_Float16)b[i];
}

// ---------------- Episode layout (sequential per batch column) ---------------

__global__ void layout_kernel(const float* __restrict__ dones,
                              int* __restrict__ src_t, int* __restrict__ cnt) {
  int b = threadIdx.x;
  if (b >= B_DIM) return;
  int* my = src_t + (size_t)b * K_EP * LM;
  for (int i = 0; i < K_EP * LM; i++) my[i] = -1;
  int ep = 0, start = 0;
  for (int t = 0; t < T_DIM; t++) {
    if (t > 0 && dones[(size_t)(t - 1) * B_DIM + b] > 0.5f) { ep++; start = t; }
    int pos = t - start;
    if (ep < K_EP && pos < LM) my[ep * LM + pos] = t;   // .at[].set(mode='drop')
  }
  for (int k = 0; k < K_EP; k++) {
    int c = 0;
    for (int p = 0; p < LM; p++) c += (my[k * LM + p] >= 0) ? 1 : 0;
    cnt[b * K_EP + k] = c;
  }
}

// ---------------- KL loss (deterministic two-phase reduction) ----------------

__global__ void kl_partial_kernel(const float* __restrict__ lm,
                                  const float* __restrict__ lv,
                                  const float* __restrict__ lmt,
                                  const float* __restrict__ lvt,
                                  float* __restrict__ partials) {
  __shared__ float red[256];
  int idx = blockIdx.x * 256 + threadIdx.x;  // over T*B
  int t = idx / B_DIM, b = idx % B_DIM;
  float sumS = 0.f, sumE = 0.f, sumExp = 0.f, sumQ = 0.f;
  for (int j = 0; j < 64; j++) {
    const float* M = (j < 32) ? lm : lmt;
    const float* V = (j < 32) ? lv : lvt;
    size_t cCur = ((size_t)t * B_DIM + b) * 32 + (j & 31);
    float mu = M[cCur];
    float logE = V[cCur];
    float m0 = 0.f, logS = 0.f;
    if (t > 0) {
      size_t cPrev = ((size_t)(t - 1) * B_DIM + b) * 32 + (j & 31);
      m0 = M[cPrev];
      logS = V[cPrev];
    }
    sumS += logS;
    sumE += logE;
    sumExp += expf(logE - logS);
    float d = m0 - mu;
    sumQ += d * d * expf(-logS);
  }
  red[threadIdx.x] = 0.5f * (sumS - sumE - 64.f + sumExp + sumQ);
  __syncthreads();
  for (int s = 128; s > 0; s >>= 1) {
    if (threadIdx.x < s) red[threadIdx.x] += red[threadIdx.x + s];
    __syncthreads();
  }
  if (threadIdx.x == 0) partials[blockIdx.x] = red[0];
}

__global__ void kl_reduce_kernel(const float* __restrict__ partials, int n,
                                 float* __restrict__ out0) {
  __shared__ float red[256];
  float s = 0.f;
  for (int i = threadIdx.x; i < n; i += 256) s += partials[i];
  red[threadIdx.x] = s;
  __syncthreads();
  for (int st = 128; st > 0; st >>= 1) {
    if (threadIdx.x < st) red[threadIdx.x] += red[threadIdx.x + st];
    __syncthreads();
  }
  if (threadIdx.x == 0) out0[0] = red[0];
}

// ---------------- Phase 1: embeddings + gi precompute via WMMA ---------------
// 16 rows per block, 8 waves. Fuses se, ce, sa, hid AND the non-recurrent
// GRU input projection gi = sa @ Wi + bi (so the sequential scan only runs
// the truly recurrent gh = h @ Wh).

__global__ __launch_bounds__(256) void embed_kernel(
    const float* __restrict__ state, const float* __restrict__ charac,
    const float* __restrict__ mental,
    const _Float16* __restrict__ WseF, const float* __restrict__ b_se,
    const _Float16* __restrict__ WceF, const float* __restrict__ b_ce,
    const _Float16* __restrict__ WsaF, const float* __restrict__ b_sa,
    const _Float16* __restrict__ WheF, const float* __restrict__ b_h,
    const _Float16* __restrict__ WiF,  const float* __restrict__ bi,
    _Float16* __restrict__ giBuf, float* __restrict__ hidBuf) {
  __shared__ _Float16 sA[16 * S_DIM];              // state rows
  __shared__ _Float16 chA[16 * C_DIM];             // character rows
  __shared__ _Float16 seca[16 * (E_S + E_C)];      // concat(se, ce)
  __shared__ _Float16 cem[16 * (E_C + M_DIM)];     // concat(ce, mental)
  __shared__ _Float16 saL[16 * H_DIM];             // sa rows (A for gi GEMM)
  int tid = threadIdx.x, lane = tid & 31, wave = tid >> 5;
  size_t r0 = (size_t)blockIdx.x * 16;

  for (int i = tid; i < 16 * S_DIM; i += 256) {
    int m = i >> 7, j = i & 127;
    sA[m * S_DIM + j] = (_Float16)state[(r0 + m) * S_DIM + j];
  }
  for (int i = tid; i < 16 * C_DIM; i += 256) {
    int m = i >> 5, j = i & 31;
    chA[m * C_DIM + j] = (_Float16)charac[(r0 + m) * C_DIM + j];
  }
  for (int i = tid; i < 16 * M_DIM; i += 256) {
    int m = i >> 5, j = i & 31;
    cem[m * (E_C + M_DIM) + E_C + j] = (_Float16)mental[(r0 + m) * M_DIM + j];
  }
  __syncthreads();

  {  // se = relu(state @ W_se + b_se): N=128 (8 tiles, wave w -> nt=w), K=128
    int nt = wave;
    int ncol = nt * 16 + (lane & 15);
    v8f acc = splat8(b_se[ncol]);
#pragma unroll
    for (int kt = 0; kt < 4; kt++) {
      v16h a = load_a_f16(sA, S_DIM, kt, lane);
      acc = wmma16(a, load_b_frag(WseF, 8, kt, nt, lane), acc);
    }
#pragma unroll
    for (int v = 0; v < 8; v++) {
      int m = v + 8 * (lane >> 4);
      float x = acc[v];
      seca[m * (E_S + E_C) + ncol] = (_Float16)(x > 0.f ? x : 0.f);
    }
  }
  if (wave < 4) {  // ce = relu(char @ W_ce + b_ce): N=64 (4 tiles), K=32
    int nt = wave;
    int ncol = nt * 16 + (lane & 15);
    v8f acc = splat8(b_ce[ncol]);
    v16h a = load_a_f16(chA, C_DIM, 0, lane);
    acc = wmma16(a, load_b_frag(WceF, 4, 0, nt, lane), acc);
#pragma unroll
    for (int v = 0; v < 8; v++) {
      int m = v + 8 * (lane >> 4);
      float x = acc[v];
      _Float16 hx = (_Float16)(x > 0.f ? x : 0.f);
      seca[m * (E_S + E_C) + E_S + ncol] = hx;
      cem[m * (E_C + M_DIM) + ncol] = hx;
    }
  }
  __syncthreads();

#pragma unroll
  for (int u = 0; u < 2; u++) {  // sa: N=256 (16 tiles, 2/wave), K=192 -> LDS
    int nt = 2 * wave + u;
    int ncol = nt * 16 + (lane & 15);
    v8f acc = splat8(b_sa[ncol]);
#pragma unroll
    for (int kt = 0; kt < 6; kt++) {
      v16h a = load_a_f16(seca, E_S + E_C, kt, lane);
      acc = wmma16(a, load_b_frag(WsaF, 16, kt, nt, lane), acc);
    }
#pragma unroll
    for (int v = 0; v < 8; v++) {
      int m = v + 8 * (lane >> 4);
      saL[m * H_DIM + ncol] = (_Float16)acc[v];
    }
  }
#pragma unroll
  for (int u = 0; u < 2; u++) {  // hid: N=256 (16 tiles, 2/wave), K=96
    int nt = 2 * wave + u;
    int ncol = nt * 16 + (lane & 15);
    v8f acc = splat8(b_h[ncol]);
#pragma unroll
    for (int kt = 0; kt < 3; kt++) {
      v16h a = load_a_f16(cem, E_C + M_DIM, kt, lane);
      acc = wmma16(a, load_b_frag(WheF, 16, kt, nt, lane), acc);
    }
#pragma unroll
    for (int v = 0; v < 8; v++) {
      int m = v + 8 * (lane >> 4);
      hidBuf[(r0 + m) * H_DIM + ncol] = acc[v];
    }
  }
  __syncthreads();

  {  // gi = sa @ Wi + bi: N=768 (48 tiles, 6/wave), K=256
    v8f accG[6];
#pragma unroll
    for (int q = 0; q < 6; q++) {
      int ncol = (wave * 6 + q) * 16 + (lane & 15);
      accG[q] = splat8(bi[ncol]);
    }
#pragma unroll
    for (int kt = 0; kt < 8; kt++) {
      v16h a = load_a_f16(saL, H_DIM, kt, lane);
#pragma unroll
      for (int q = 0; q < 6; q++) {
        int nt = wave * 6 + q;
        accG[q] = wmma16(a, load_b_frag(WiF, 48, kt, nt, lane), accG[q]);
      }
    }
#pragma unroll
    for (int q = 0; q < 6; q++) {
      int ncol = (wave * 6 + q) * 16 + (lane & 15);
#pragma unroll
      for (int v = 0; v < 8; v++) {
        int m = v + 8 * (lane >> 4);
        giBuf[(r0 + m) * G_DIM + ncol] = (_Float16)accG[q][v];
      }
    }
  }
}

// ---------------- Phase 2: sequential GRU scan (gh only) ---------------------
// One block per batch b; 16 rows = 16 episodes; 8 waves; wave w owns H-columns
// [32w, 32w+32) of all three gates so gating is wave-local. gi is gathered
// (precomputed); masked slots read gi = bi (x==0 in the reference).

__global__ __launch_bounds__(256) void scan_kernel(
    const _Float16* __restrict__ giBuf, const float* __restrict__ hidBuf,
    const float* __restrict__ dones, const int* __restrict__ src_t,
    const _Float16* __restrict__ bi16,
    const _Float16* __restrict__ WhF, const float* __restrict__ bh,
    const _Float16* __restrict__ WoF, const float* __restrict__ b_out,
    float* __restrict__ logitsBuf) {
  __shared__ float    h32s[16 * H_DIM];   // canonical hidden state (f32)
  __shared__ _Float16 h16s[16 * H_DIM];   // f16 shadow for A-fragments
  __shared__ _Float16 giL[16 * G_DIM];    // gathered gi rows (f16)
  int tid = threadIdx.x, lane = tid & 31, wave = tid >> 5;
  int b = blockIdx.x;
  const int* srcB = src_t + (size_t)b * K_EP * LM;

  {  // h0 = k_hid[:, :, 0, :]
    int k = tid >> 4, j0 = tid & 15;
    int t0 = srcB[k * LM + 0];
#pragma unroll
    for (int i = 0; i < 16; i++) {
      int j = j0 + 16 * i;
      h32s[k * H_DIM + j] =
          (t0 >= 0) ? hidBuf[((size_t)t0 * B_DIM + b) * H_DIM + j] : 0.f;
    }
  }
  __syncthreads();

  for (int p = 0; p < LM; p++) {
    {  // stage 1: gather gi row, done-reset, rebuild f16 shadow
      int k = tid >> 4, l16 = tid & 15;
      int t = srcB[k * LM + p];
      const unsigned int* src32 =
          (t >= 0)
              ? (const unsigned int*)(giBuf + ((size_t)t * B_DIM + b) * G_DIM)
              : (const unsigned int*)bi16;
      unsigned int* dst32 = (unsigned int*)(giL + k * G_DIM);
#pragma unroll
      for (int i = 0; i < 24; i++) {  // 384 dwords per row / 16 threads
        int j = l16 + 16 * i;
        dst32[j] = src32[j];
      }
      // prefetch next step's gi row toward this WGP while the GEMM runs
      if (p + 1 < LM) {
        int tn = srcB[k * LM + p + 1];
        if (tn >= 0) {
          const char* pn =
              (const char*)(giBuf + ((size_t)tn * B_DIM + b) * G_DIM) +
              l16 * 96;  // this thread's 96-byte chunk
          __builtin_prefetch(pn, 0, 1);
        }
      }
      size_t rowoff = 0;
      bool reset = false;
      if (t >= 0) {
        rowoff = ((size_t)t * B_DIM + b) * H_DIM;
        reset = dones[(size_t)t * B_DIM + b] > 0.5f;
      }
#pragma unroll
      for (int i = 0; i < 16; i++) {
        int j = l16 + 16 * i;
        float hv = h32s[k * H_DIM + j];
        if (reset) { hv = hidBuf[rowoff + j]; h32s[k * H_DIM + j] = hv; }
        h16s[k * H_DIM + j] = (_Float16)hv;
      }
    }
    __syncthreads();

    // stage 2: gh = h @ Wh + bh for this wave's 6 N-tiles, then gate
    v8f accH[3][2];
#pragma unroll
    for (int g = 0; g < 3; g++)
#pragma unroll
      for (int u = 0; u < 2; u++) {
        int ncol = (g * 16 + 2 * wave + u) * 16 + (lane & 15);
        accH[g][u] = splat8(bh[ncol]);
      }
#pragma unroll
    for (int kt = 0; kt < 8; kt++) {
      v16h a_h = load_a_f16(h16s, H_DIM, kt, lane);
#pragma unroll
      for (int g = 0; g < 3; g++)
#pragma unroll
        for (int u = 0; u < 2; u++) {
          int nt = g * 16 + 2 * wave + u;
          accH[g][u] = wmma16(a_h, load_b_frag(WhF, 48, kt, nt, lane), accH[g][u]);
        }
    }
#pragma unroll
    for (int u = 0; u < 2; u++) {
      int n = (2 * wave + u) * 16 + (lane & 15);  // H column (wave-owned)
#pragma unroll
      for (int v = 0; v < 8; v++) {
        int m = v + 8 * (lane >> 4);
        float ir = (float)giL[m * G_DIM + n];
        float iz = (float)giL[m * G_DIM + 256 + n];
        float in_ = (float)giL[m * G_DIM + 512 + n];
        float r = 1.f / (1.f + expf(-(ir + accH[0][u][v])));
        float z = 1.f / (1.f + expf(-(iz + accH[1][u][v])));
        float nn = tanhf(in_ + r * accH[2][u][v]);
        float hold = h32s[m * H_DIM + n];
        h32s[m * H_DIM + n] = (1.f - z) * nn + z * hold;
      }
    }
    __syncthreads();

    // stage 3: logits = h_new @ W_out + b_out (wave 0), masked, K=256
    if (wave == 0) {
      int acol = lane & 15;
      v8f acc = splat8(b_out[acol]);
#pragma unroll
      for (int kt = 0; kt < 8; kt++) {
        v16h a = load_a_f32(h32s, H_DIM, kt, lane);
        acc = wmma16(a, load_b_frag(WoF, 1, kt, 0, lane), acc);
      }
#pragma unroll
      for (int v = 0; v < 8; v++) {
        int m = v + 8 * (lane >> 4);
        int t = srcB[m * LM + p];
        logitsBuf[(((size_t)b * K_EP + m) * LM + p) * A_DIM + acol] =
            (t >= 0) ? acc[v] : 0.f;
      }
    }
    __syncthreads();
  }
}

// ---------------- Finalize: shift-to-front, sum over K, log_softmax ----------

__global__ void finalize_kernel(const float* __restrict__ logitsBuf,
                                const int* __restrict__ cnt,
                                const int* __restrict__ actions,
                                float* __restrict__ out) {
  int p = blockIdx.x;
  int b = threadIdx.x;  // 32 threads = one wave
  float s[A_DIM];
#pragma unroll
  for (int a = 0; a < A_DIM; a++) s[a] = 0.f;
  for (int k = 0; k < K_EP; k++) {
    int c = cnt[b * K_EP + k];
    int sp = p - (LM - c);
    if (sp >= 0) {
      const float* L = logitsBuf + (((size_t)b * K_EP + k) * LM + sp) * A_DIM;
#pragma unroll
      for (int a = 0; a < A_DIM; a++) s[a] += L[a];
    }
  }
  float mx = s[0];
#pragma unroll
  for (int a = 1; a < A_DIM; a++) mx = fmaxf(mx, s[a]);
  float se = 0.f;
#pragma unroll
  for (int a = 0; a < A_DIM; a++) se += expf(s[a] - mx);
  float lse = mx + logf(se);
  int act = actions[p * B_DIM + b];
  float logp = s[act] - lse;
#pragma unroll
  for (int m = 16; m > 0; m >>= 1) logp += __shfl_xor(logp, m, 32);
  if (b == 0) out[1 + p] = logp;
}

// ---------------- Host launcher ----------------------------------------------

extern "C" void kernel_launch(void* const* d_in, const int* in_sizes, int n_in,
                              void* d_out, int out_size, void* d_ws,
                              size_t ws_size, hipStream_t stream) {
  const float* state  = (const float*)d_in[0];
  const float* lm     = (const float*)d_in[1];
  const float* lv     = (const float*)d_in[2];
  const float* lmt    = (const float*)d_in[3];
  const float* lvt    = (const float*)d_in[4];
  const float* charac = (const float*)d_in[5];
  const float* mental = (const float*)d_in[6];
  const int*   pact   = (const int*)d_in[7];
  const float* dones  = (const float*)d_in[8];
  const float* W_se   = (const float*)d_in[9];
  const float* b_se   = (const float*)d_in[10];
  const float* W_ce   = (const float*)d_in[11];
  const float* b_ce   = (const float*)d_in[12];
  const float* W_sa   = (const float*)d_in[13];
  const float* b_sa   = (const float*)d_in[14];
  const float* W_h    = (const float*)d_in[15];
  const float* b_h    = (const float*)d_in[16];
  const float* Wi     = (const float*)d_in[17];
  const float* bi     = (const float*)d_in[18];
  const float* Wh     = (const float*)d_in[19];
  const float* bh     = (const float*)d_in[20];
  const float* W_out  = (const float*)d_in[21];
  const float* b_out  = (const float*)d_in[22];
  float* out = (float*)d_out;
  (void)in_sizes; (void)n_in; (void)out_size; (void)ws_size;

  char* ws = (char*)d_ws;
  size_t off = 0;
  auto alloc = [&](size_t bytes) -> char* {
    char* p = ws + off;
    off += (bytes + 255) & ~(size_t)255;
    return p;
  };
  _Float16* giBuf   = (_Float16*)alloc((size_t)ROWS * G_DIM * 2);
  float* hidBuf     = (float*)alloc((size_t)ROWS * H_DIM * 4);
  float* logitsBuf  = (float*)alloc((size_t)B_DIM * K_EP * LM * A_DIM * 4);
  int*   srcT       = (int*)alloc((size_t)B_DIM * K_EP * LM * 4);
  int*   cnt        = (int*)alloc(B_DIM * K_EP * 4);
  float* partials   = (float*)alloc((ROWS / 256) * 4);
  _Float16* bi16 = (_Float16*)alloc((size_t)G_DIM * 2);
  _Float16* WseF = (_Float16*)alloc((size_t)4 * 8 * 512 * 2);
  _Float16* WceF = (_Float16*)alloc((size_t)1 * 4 * 512 * 2);
  _Float16* WsaF = (_Float16*)alloc((size_t)6 * 16 * 512 * 2);
  _Float16* WheF = (_Float16*)alloc((size_t)3 * 16 * 512 * 2);
  _Float16* WiF  = (_Float16*)alloc((size_t)8 * 48 * 512 * 2);
  _Float16* WhF  = (_Float16*)alloc((size_t)8 * 48 * 512 * 2);
  _Float16* WoF  = (_Float16*)alloc((size_t)8 * 1 * 512 * 2);

  auto launch_swz = [&](const float* W, int K, int N, _Float16* dst) {
    int KT = (K + 31) / 32, NT = (N + 15) / 16;
    int total = KT * NT * 32;
    swizzle_b_kernel<<<(total + 255) / 256, 256, 0, stream>>>(W, K, N, dst, KT, NT);
  };
  launch_swz(W_se, 128, 128, WseF);
  launch_swz(W_ce, 32, 64, WceF);
  launch_swz(W_sa, 192, 256, WsaF);
  launch_swz(W_h, 96, 256, WheF);
  launch_swz(Wi, 256, 768, WiF);
  launch_swz(Wh, 256, 768, WhF);
  launch_swz(W_out, 256, 16, WoF);
  bias16_kernel<<<(G_DIM + 255) / 256, 256, 0, stream>>>(bi, G_DIM, bi16);

  layout_kernel<<<1, 32, 0, stream>>>(dones, srcT, cnt);
  kl_partial_kernel<<<ROWS / 256, 256, 0, stream>>>(lm, lv, lmt, lvt, partials);
  kl_reduce_kernel<<<1, 256, 0, stream>>>(partials, ROWS / 256, out);
  embed_kernel<<<ROWS / 16, 256, 0, stream>>>(state, charac, mental, WseF, b_se,
                                              WceF, b_ce, WsaF, b_sa, WheF, b_h,
                                              WiF, bi, giBuf, hidBuf);
  scan_kernel<<<B_DIM, 256, 0, stream>>>(giBuf, hidBuf, dones, srcT, bi16, WhF,
                                         bh, WoF, b_out, logitsBuf);
  finalize_kernel<<<LM, 32, 0, stream>>>(logitsBuf, cnt, pact, out);
}
